// PyramidDeformableAlignment_47863115546744
// MI455X (gfx1250) — matrix operand order repaired and checked by reading, not compiled, not run on hardware
//
#include <hip/hip_runtime.h>
#include <math.h>

// ============================================================================
// MI455X (gfx1250) implementation of Pyramid Deformable Alignment.
// All conv / deform-conv contractions are lowered to f16 WMMA GEMMs
// (v_wmma_f32_16x16x32_f16, wave32), with im2col panels written in WMMA
// A-fragment order and weights pre-packed in B-fragment order so the GEMM
// inner loop is pure b128 loads + WMMA. Strips of 4096 pixels keep every
// A panel (<= 51 MB) resident in the 192 MB L2.
//
// GEMM: each wave computes a 32x16 tile (2 WMMAs share one B fragment);
// the block's B panel is staged into LDS once via the gfx1250 async
// global->LDS path (GLOBAL_LOAD_ASYNC_TO_LDS_B128 + s_wait_asynccnt) and
// read back with ds_load_b128.
// ============================================================================

typedef __attribute__((ext_vector_type(16))) _Float16 v16h;
typedef __attribute__((ext_vector_type(8)))  float    v8f;
typedef __attribute__((ext_vector_type(4)))  unsigned int v4u;
typedef __attribute__((ext_vector_type(4)))  int      v4i;

#define MSTRIP 4096   // pixels per im2col strip (multiple of 128)
#define BN     8      // B*(T-1) = 2*4 frames processed
#define LDS_B_CAP (160 * 1024)

#ifndef HAVE_ASYNC_LDS
#if defined(__has_builtin)
#if __has_builtin(__builtin_amdgcn_global_load_async_to_lds_b128) && \
    __has_builtin(__builtin_amdgcn_s_wait_asynccnt)
#define HAVE_ASYNC_LDS 1
#endif
#endif
#endif
#ifndef HAVE_ASYNC_LDS
#define HAVE_ASYNC_LDS 0
#endif

static inline int cdiv(int a, int b) { return (a + b - 1) / b; }

// ---------------------------------------------------------------------------
// Weight packing: w[O][Cin][kh][kw] (f32) -> f16 B-fragments.
// Storage order: [chunk = K/32][ntile = O/16][lane 0..31][i 0..15]
// Fragment semantics (16-bit B 32x16): n = lane&15,
//   K = chunk*32 + (i<8 ? (lane>>4)*8 + i : 16 + (lane>>4)*8 + (i-8))
// ---------------------------------------------------------------------------
__global__ void pack_w_k(const float* __restrict__ w, _Float16* __restrict__ bp,
                         int Cin, int area, int O, int total) {
  int tid = blockIdx.x * blockDim.x + threadIdx.x;
  if (tid >= total) return;
  int i     = tid & 15;
  int lane  = (tid >> 4) & 31;
  int rest  = tid >> 9;
  int Nt    = O >> 4;
  int nt    = rest % Nt;
  int chunk = rest / Nt;
  int kkG   = chunk * 32 + ((i < 8) ? ((lane >> 4) * 8 + i) : (16 + (lane >> 4) * 8 + (i - 8)));
  int n     = nt * 16 + (lane & 15);
  int ci    = kkG / area;
  int r     = kkG - ci * area;
  bp[tid] = (_Float16)w[((size_t)n * Cin + ci) * area + r];
}

// ---------------------------------------------------------------------------
// im2col into f16 A-fragments for one image strip.
// Input may be a virtual channel-concat of two tensors (s0: ci<Cin0, s1 rest).
// Storage order: [mtile][chunk][lane][i]
// Fragment semantics (16-bit A 16x32): m = lane&15,
//   K = chunk*32 + (i<8 ? (lane>>4)*8 + i : 16 + (lane>>4)*8 + (i-8))
// ---------------------------------------------------------------------------
__global__ void im2col_k(const float* __restrict__ s0, const float* __restrict__ s1,
                         int Cin0, int H, int W, int kh, int pad, int mStart,
                         int K32, _Float16* __restrict__ Ap, int total) {
  int tid = blockIdx.x * blockDim.x + threadIdx.x;
  if (tid >= total) return;
  int i     = tid & 15;
  int lane  = (tid >> 4) & 31;
  int rest  = tid >> 9;
  int chunk = rest % K32;
  int mtile = rest / K32;
  int m = mStart + mtile * 16 + (lane & 15);
  int y = m / W, x = m - y * W;
  int khalf = lane >> 4;
  int kkG = chunk * 32 + ((i < 8) ? (khalf * 8 + i) : (16 + khalf * 8 + (i - 8)));
  int area = kh * kh;
  int ci = kkG / area;
  int r  = kkG - ci * area;
  int ky = r / kh, kx = r - ky * kh;
  int yy = y + ky - pad, xx = x + kx - pad;
  float v = 0.0f;
  if (yy >= 0 && yy < H && xx >= 0 && xx < W) {
    const float* s = (ci < Cin0) ? (s0 + (size_t)ci * H * W)
                                 : (s1 + (size_t)(ci - Cin0) * H * W);
    v = s[yy * W + xx];
  }
  Ap[tid] = (_Float16)v;
}

// ---------------------------------------------------------------------------
// Deformable im2col (DCNv1): bilinear sample at learned fractional offsets,
// zero outside bounds. Cin=64, dg=8 (Cg=8), 3x3 kernel, 144 offset channels
// laid out [dg][9][2(dy,dx)][H][W].  Same A-fragment storage as im2col_k.
// ---------------------------------------------------------------------------
__global__ void dim2col_k(const float* __restrict__ s, const float* __restrict__ off,
                          int H, int W, int mStart, int K32,
                          _Float16* __restrict__ Ap, int total) {
  int tid = blockIdx.x * blockDim.x + threadIdx.x;
  if (tid >= total) return;
  int i     = tid & 15;
  int lane  = (tid >> 4) & 31;
  int rest  = tid >> 9;
  int chunk = rest % K32;
  int mtile = rest / K32;
  int HW = H * W;
  int m = mStart + mtile * 16 + (lane & 15);
  int y = m / W, x = m - y * W;
  int khalf = lane >> 4;
  int kkG = chunk * 32 + ((i < 8) ? (khalf * 8 + i) : (16 + khalf * 8 + (i - 8)));
  int ci = kkG / 9;
  int r  = kkG - ci * 9;
  int g  = ci >> 3;                       // Cg = 64/8 = 8
  int oc = (g * 9 + r) * 2;
  float dy = off[(size_t)oc * HW + m];
  float dx = off[(size_t)(oc + 1) * HW + m];
  float py = (float)y + (float)(r / 3 - 1) + dy;
  float px = (float)x + (float)(r % 3 - 1) + dx;
  float fy = floorf(py), fx = floorf(px);
  int   iy = (int)fy,    ix = (int)fx;
  float wy = py - fy,    wx = px - fx;
  const float* sp = s + (size_t)ci * HW;
  float w00 = (1.0f - wy) * (1.0f - wx), w01 = (1.0f - wy) * wx;
  float w10 = wy * (1.0f - wx),          w11 = wy * wx;
  float acc = 0.0f;
  if (iy >= 0 && iy < H) {
    if (ix     >= 0 && ix     < W) acc += w00 * sp[iy * W + ix];
    if (ix + 1 >= 0 && ix + 1 < W) acc += w01 * sp[iy * W + ix + 1];
  }
  if (iy + 1 >= 0 && iy + 1 < H) {
    if (ix     >= 0 && ix     < W) acc += w10 * sp[(iy + 1) * W + ix];
    if (ix + 1 >= 0 && ix + 1 < W) acc += w11 * sp[(iy + 1) * W + ix + 1];
  }
  Ap[tid] = (_Float16)acc;
}

// ---------------------------------------------------------------------------
// WMMA GEMM: C[M,N] f32 = A[M,K] f16 * B[K,N] f16 (+bias), NCHW scatter.
// Block = 128 threads = 4 wave32s. Each wave owns a 32x16 tile (2 M-tiles
// share each B fragment). If useLds, the block's whole B panel (K32 x 1KB)
// is staged into LDS once via async global->LDS and the inner loop reads B
// with ds_load_b128.
// ---------------------------------------------------------------------------
__global__ void wmma_gemm_k(const _Float16* __restrict__ Ap,
                            const _Float16* __restrict__ Bp,
                            const float* __restrict__ bias,
                            float* __restrict__ out,
                            int K32, int Ntiles, int HW, int mStart, int useLds) {
  extern __shared__ _Float16 ldsB[];
  int lane  = threadIdx.x & 31;
  int wave  = threadIdx.x >> 5;
  int ntile = blockIdx.y;

  if (useLds) {
    // Cooperative stage of the N-tile's B panel: K32 chunks x 512 halves.
    for (int idx = threadIdx.x; idx < K32 * 64; idx += 128) {
      int ch = idx >> 6, t = idx & 63;
      const _Float16* src = Bp + ((size_t)ch * Ntiles + ntile) * 512 + t * 8;
      _Float16* dst = ldsB + ch * 512 + t * 8;
#if HAVE_ASYNC_LDS
      // Builtin proto (from toolchain): (v4i AS1* src, v4i AS3* dst, Imm off, Imm cpol)
      const __attribute__((address_space(1))) void* gsrc =
          (const __attribute__((address_space(1))) void*)src;
      __attribute__((address_space(3))) void* ldst =
          (__attribute__((address_space(3))) void*)dst;
      __builtin_amdgcn_global_load_async_to_lds_b128(
          (__attribute__((address_space(1))) v4i*)gsrc,
          (__attribute__((address_space(3))) v4i*)ldst, 0, 0);
#else
      *(v4u*)dst = *(const v4u*)src;
#endif
    }
#if HAVE_ASYNC_LDS
    __builtin_amdgcn_s_wait_asynccnt(0);
#endif
  }
  __syncthreads();

  int mt0 = (blockIdx.x * 4 + wave) * 2;
  const _Float16* aP0 = Ap + (size_t)mt0 * K32 * 512 + lane * 16;
  const _Float16* aP1 = aP0 + (size_t)K32 * 512;
  v8f acc0 = {};
  v8f acc1 = {};

  if (useLds) {
    const _Float16* bL = ldsB + lane * 16;
    for (int c = 0; c < K32; ++c) {
      v16h a0 = *(const v16h*)aP0;
      v16h a1 = *(const v16h*)aP1;
      v16h b  = *(const v16h*)bL;
      __builtin_prefetch(aP0 + 4 * 512, 0, 1);
      __builtin_prefetch(aP1 + 4 * 512, 0, 1);
      acc0 = __builtin_amdgcn_wmma_f32_16x16x32_f16(false, a0, false, b,
                                                    (short)0, acc0, false, false);
      acc1 = __builtin_amdgcn_wmma_f32_16x16x32_f16(false, a1, false, b,
                                                    (short)0, acc1, false, false);
      aP0 += 512; aP1 += 512; bL += 512;
    }
  } else {
    const _Float16* bG = Bp + (size_t)ntile * 512 + lane * 16;
    const size_t bStride = (size_t)Ntiles * 512;
    for (int c = 0; c < K32; ++c) {
      v16h a0 = *(const v16h*)aP0;
      v16h a1 = *(const v16h*)aP1;
      v16h b  = *(const v16h*)bG;
      __builtin_prefetch(aP0 + 4 * 512, 0, 1);
      __builtin_prefetch(aP1 + 4 * 512, 0, 1);
      acc0 = __builtin_amdgcn_wmma_f32_16x16x32_f16(false, a0, false, b,
                                                    (short)0, acc0, false, false);
      acc1 = __builtin_amdgcn_wmma_f32_16x16x32_f16(false, a1, false, b,
                                                    (short)0, acc1, false, false);
      aP0 += 512; aP1 += 512; bG += bStride;
    }
  }

  // C layout: n = lane&15, M = r + 8*(lane>>4)
  int n = ntile * 16 + (lane & 15);
  float bv = bias ? bias[n] : 0.0f;
  int mBase = mStart + mt0 * 16 + ((lane >> 4) << 3);
  float* o0 = out + (size_t)n * HW + mBase;
  float* o1 = o0 + 16;
#pragma unroll
  for (int r = 0; r < 8; ++r) {
    o0[r] = acc0[r] + bv;
    o1[r] = acc1[r] + bv;
  }
}

// ---------------------------------------------------------------------------
// Elementwise / data-movement kernels
// ---------------------------------------------------------------------------
__global__ void pool2x2_k(const float* __restrict__ in, float* __restrict__ out,
                          int H, int W, int total) {
  int tid = blockIdx.x * blockDim.x + threadIdx.x;
  if (tid >= total) return;
  int Wo = W >> 1, Ho = H >> 1;
  int x = tid % Wo;
  int y = (tid / Wo) % Ho;
  int n = tid / (Wo * Ho);
  const float* p = in + ((size_t)n * H + 2 * y) * W + 2 * x;
  out[tid] = 0.25f * (p[0] + p[1] + p[W] + p[W + 1]);
}

__global__ void gather_k(const float* __restrict__ X, float* __restrict__ nb,
                         float* __restrict__ cb, int T, int tc, int CHW, int total) {
  int tid = blockIdx.x * blockDim.x + threadIdx.x;
  if (tid >= total) return;
  int i = tid % CHW;
  int j = tid / CHW;
  int b = j / (T - 1);
  int f = j % (T - 1);
  int t = (f < tc) ? f : f + 1;
  nb[tid] = X[(size_t)(b * T + t) * CHW + i];
  cb[tid] = X[(size_t)(b * T + tc) * CHW + i];
}

__global__ void upsample2_k(const float* __restrict__ in, float* __restrict__ out,
                            int H, int W, float scale, int total) {
  int tid = blockIdx.x * blockDim.x + threadIdx.x;
  if (tid >= total) return;
  int Wo = W * 2, Ho = H * 2;
  int ox = tid % Wo;
  int oy = (tid / Wo) % Ho;
  int n  = tid / (Wo * Ho);
  float fy = (oy + 0.5f) * 0.5f - 0.5f;
  float fx = (ox + 0.5f) * 0.5f - 0.5f;
  float y0 = floorf(fy), x0 = floorf(fx);
  float wy = fy - y0,    wx = fx - x0;
  int iy0 = (int)y0, ix0 = (int)x0;
  int iy1 = iy0 + 1, ix1 = ix0 + 1;
  iy0 = iy0 < 0 ? 0 : (iy0 > H - 1 ? H - 1 : iy0);
  iy1 = iy1 < 0 ? 0 : (iy1 > H - 1 ? H - 1 : iy1);
  ix0 = ix0 < 0 ? 0 : (ix0 > W - 1 ? W - 1 : ix0);
  ix1 = ix1 < 0 ? 0 : (ix1 > W - 1 ? W - 1 : ix1);
  const float* p = in + (size_t)n * H * W;
  float v = (1.0f - wy) * ((1.0f - wx) * p[iy0 * W + ix0] + wx * p[iy0 * W + ix1]) +
            wy          * ((1.0f - wx) * p[iy1 * W + ix0] + wx * p[iy1 * W + ix1]);
  out[tid] = v * scale;
}

__global__ void add_k(float* __restrict__ a, const float* __restrict__ b, int total) {
  int tid = blockIdx.x * blockDim.x + threadIdx.x;
  if (tid >= total) return;
  a[tid] += b[tid];
}

__global__ void assemble_k(const float* __restrict__ X, const float* __restrict__ A,
                           float* __restrict__ out, int T, int tc, int CHW, int total) {
  int tid = blockIdx.x * blockDim.x + threadIdx.x;
  if (tid >= total) return;
  int i  = tid % CHW;
  int bt = tid / CHW;
  int t  = bt % T;
  int b  = bt / T;
  out[tid] = (t == tc) ? X[tid]
                       : A[(size_t)(b * (T - 1) + (t < tc ? t : t - 1)) * CHW + i];
}

// ---------------------------------------------------------------------------
// Host-side orchestration helpers (graph-capture safe: pure enqueues)
// ---------------------------------------------------------------------------
namespace {

struct Arena {
  char*  base;
  size_t off;
  size_t cap;
};

static void* arena_alloc(Arena& a, size_t bytes) {
  size_t al = (a.off + 255) & ~(size_t)255;
  a.off = al + bytes;
  return a.base + al;   // assumes ws_size is sufficient (~600 MB)
}
static float*    allocF(Arena& a, size_t n) { return (float*)arena_alloc(a, n * 4); }
static _Float16* allocH(Arena& a, size_t n) { return (_Float16*)arena_alloc(a, n * 2); }

static _Float16* pack(hipStream_t st, Arena& ar, const float* w, int Cin, int area, int O) {
  size_t total = (size_t)Cin * area * O;
  _Float16* bp = allocH(ar, total);
  pack_w_k<<<cdiv((int)total, 256), 256, 0, st>>>(w, bp, Cin, area, O, (int)total);
  return bp;
}

static void gemm_launch(hipStream_t st, const _Float16* panel, const _Float16* Bp,
                        const float* bias, float* ob, int K32, int O, int HW,
                        int mS, int mC) {
  int ldsBytes = K32 * 1024;
  int useLds = (ldsBytes <= LDS_B_CAP) ? 1 : 0;
  size_t sh = useLds ? (size_t)ldsBytes : 0;
  dim3 g(mC / 128, O / 16);
  wmma_gemm_k<<<g, dim3(128), sh, st>>>(panel, Bp, bias, ob, K32, O / 16, HW, mS,
                                        useLds);
}

static void conv_layer(hipStream_t st, const float* in0, const float* in1, int Cin0,
                       int Cin, int H, int W, int kh, int pad, const _Float16* Bp,
                       const float* bias, int O, float* out, _Float16* panel) {
  int HW = H * W, K = Cin * kh * kh, K32 = K / 32;
  for (int b = 0; b < BN; ++b) {
    const float* s0 = in0 + (size_t)b * Cin0 * HW;
    const float* s1 = in1 ? (in1 + (size_t)b * (Cin - Cin0) * HW) : nullptr;
    float* ob = out + (size_t)b * O * HW;
    for (int mS = 0; mS < HW; mS += MSTRIP) {
      int mC = (HW - mS < MSTRIP) ? (HW - mS) : MSTRIP;
      int total = mC * K;
      im2col_k<<<cdiv(total, 256), 256, 0, st>>>(s0, s1, Cin0, H, W, kh, pad, mS,
                                                 K32, panel, total);
      gemm_launch(st, panel, Bp, bias, ob, K32, O, HW, mS, mC);
    }
  }
}

static void dconv_layer(hipStream_t st, const float* in, const float* off, int H, int W,
                        const _Float16* Bp, int O, float* out, _Float16* panel) {
  int HW = H * W, K = 576, K32 = 18;   // Cin=64, 3x3
  for (int b = 0; b < BN; ++b) {
    const float* s   = in  + (size_t)b * 64  * HW;
    const float* ofb = off + (size_t)b * 144 * HW;
    float* ob = out + (size_t)b * O * HW;
    for (int mS = 0; mS < HW; mS += MSTRIP) {
      int mC = (HW - mS < MSTRIP) ? (HW - mS) : MSTRIP;
      int total = mC * K;
      dim2col_k<<<cdiv(total, 256), 256, 0, st>>>(s, ofb, H, W, mS, K32, panel, total);
      gemm_launch(st, panel, Bp, nullptr, ob, K32, O, HW, mS, mC);
    }
  }
}

static void offset_branch(hipStream_t st, const float* neig, const float* cen, int H,
                          int W, int kh, int pad, const _Float16* cw, const float* cb,
                          const _Float16* d1ow, const float* d1ob, const _Float16* d1w,
                          const _Float16* d2ow, const float* d2ob, const _Float16* d2w,
                          float* outOff, float* tmpH, float* tmpH2, float* tmpOff,
                          _Float16* panel) {
  conv_layer(st, neig, cen, 64, 128, H, W, kh, pad, cw, cb, 64, tmpH, panel);
  conv_layer(st, tmpH, nullptr, 64, 64, H, W, 3, 1, d1ow, d1ob, 144, tmpOff, panel);
  dconv_layer(st, tmpH, tmpOff, H, W, d1w, 64, tmpH2, panel);
  conv_layer(st, tmpH2, nullptr, 64, 64, H, W, 3, 1, d2ow, d2ob, 144, tmpOff, panel);
  dconv_layer(st, tmpH2, tmpOff, H, W, d2w, 144, outOff, panel);
}

}  // namespace

// ---------------------------------------------------------------------------
// kernel_launch
// ---------------------------------------------------------------------------
extern "C" void kernel_launch(void* const* d_in, const int* in_sizes, int n_in,
                              void* d_out, int out_size, void* d_ws, size_t ws_size,
                              hipStream_t stream) {
  (void)in_sizes; (void)n_in; (void)out_size;

  // Inputs (setup_inputs insertion order: x, then params o1,o2,o3,al2,al1,fi,post)
  const float* X        = (const float*)d_in[0];
  const float* o1_cw    = (const float*)d_in[1];
  const float* o1_cb    = (const float*)d_in[2];
  const float* o1_d1ow  = (const float*)d_in[3];
  const float* o1_d1ob  = (const float*)d_in[4];
  const float* o1_d1w   = (const float*)d_in[5];
  const float* o1_d2ow  = (const float*)d_in[6];
  const float* o1_d2ob  = (const float*)d_in[7];
  const float* o1_d2w   = (const float*)d_in[8];
  const float* o2_cw    = (const float*)d_in[9];
  const float* o2_cb    = (const float*)d_in[10];
  const float* o2_d1ow  = (const float*)d_in[11];
  const float* o2_d1ob  = (const float*)d_in[12];
  const float* o2_d1w   = (const float*)d_in[13];
  const float* o2_d2ow  = (const float*)d_in[14];
  const float* o2_d2ob  = (const float*)d_in[15];
  const float* o2_d2w   = (const float*)d_in[16];
  const float* o3_cw    = (const float*)d_in[17];
  const float* o3_cb    = (const float*)d_in[18];
  const float* o3_d1ow  = (const float*)d_in[19];
  const float* o3_d1ob  = (const float*)d_in[20];
  const float* o3_d1w   = (const float*)d_in[21];
  const float* o3_d2ow  = (const float*)d_in[22];
  const float* o3_d2ob  = (const float*)d_in[23];
  const float* o3_d2w   = (const float*)d_in[24];
  const float* al2_w    = (const float*)d_in[25];
  const float* al1_w    = (const float*)d_in[26];
  const float* fi_w     = (const float*)d_in[27];
  const float* post_ow  = (const float*)d_in[28];
  const float* post_ob  = (const float*)d_in[29];
  const float* post_w   = (const float*)d_in[30];

  const int T = 5, tc = 2, C = 64;
  const int H1 = 128, W1 = 128, HW1 = H1 * W1;
  const int H2 = 64,  W2 = 64,  HW2 = H2 * W2;
  const int H3 = 32,  W3 = 32,  HW3 = H3 * W3;

  Arena ar{(char*)d_ws, 0, ws_size};

  // ---- pack weights to WMMA f16 fragments ----
  _Float16* P_o1cw   = pack(stream, ar, o1_cw,  128, 49, 64);
  _Float16* P_o1d1ow = pack(stream, ar, o1_d1ow, 64, 9, 144);
  _Float16* P_o1d1w  = pack(stream, ar, o1_d1w,  64, 9, 64);
  _Float16* P_o1d2ow = pack(stream, ar, o1_d2ow, 64, 9, 144);
  _Float16* P_o1d2w  = pack(stream, ar, o1_d2w,  64, 9, 144);
  _Float16* P_o2cw   = pack(stream, ar, o2_cw,  128, 25, 64);
  _Float16* P_o2d1ow = pack(stream, ar, o2_d1ow, 64, 9, 144);
  _Float16* P_o2d1w  = pack(stream, ar, o2_d1w,  64, 9, 64);
  _Float16* P_o2d2ow = pack(stream, ar, o2_d2ow, 64, 9, 144);
  _Float16* P_o2d2w  = pack(stream, ar, o2_d2w,  64, 9, 144);
  _Float16* P_o3cw   = pack(stream, ar, o3_cw,  128, 9, 64);
  _Float16* P_o3d1ow = pack(stream, ar, o3_d1ow, 64, 9, 144);
  _Float16* P_o3d1w  = pack(stream, ar, o3_d1w,  64, 9, 64);
  _Float16* P_o3d2ow = pack(stream, ar, o3_d2ow, 64, 9, 144);
  _Float16* P_o3d2w  = pack(stream, ar, o3_d2w,  64, 9, 144);
  _Float16* P_al2    = pack(stream, ar, al2_w,   64, 9, 64);
  _Float16* P_al1    = pack(stream, ar, al1_w,   64, 9, 64);
  _Float16* P_fi     = pack(stream, ar, fi_w,    64, 9, 64);
  _Float16* P_postow = pack(stream, ar, post_ow, 64, 9, 144);
  _Float16* P_postw  = pack(stream, ar, post_w,  64, 9, 64);

  // ---- activation buffers ----
  float* X2    = allocF(ar, (size_t)2 * T * C * HW2);
  float* X3    = allocF(ar, (size_t)2 * T * C * HW3);
  float* n1    = allocF(ar, (size_t)BN * C * HW1);
  float* c1    = allocF(ar, (size_t)BN * C * HW1);
  float* n2    = allocF(ar, (size_t)BN * C * HW2);
  float* c2    = allocF(ar, (size_t)BN * C * HW2);
  float* n3    = allocF(ar, (size_t)BN * C * HW3);
  float* c3    = allocF(ar, (size_t)BN * C * HW3);
  float* tmpH  = allocF(ar, (size_t)BN * C * HW1);
  float* tmpH2 = allocF(ar, (size_t)BN * C * HW1);
  float* tmpOff= allocF(ar, (size_t)BN * 144 * HW1);
  float* off3  = allocF(ar, (size_t)BN * 144 * HW3);
  float* off3u = allocF(ar, (size_t)BN * 144 * HW2);
  float* z2    = allocF(ar, (size_t)BN * C * HW2);
  float* off2  = allocF(ar, (size_t)BN * 144 * HW2);
  float* off2u = allocF(ar, (size_t)BN * 144 * HW1);
  float* z1    = allocF(ar, (size_t)BN * C * HW1);
  float* off1  = allocF(ar, (size_t)BN * 144 * HW1);
  float* a1    = allocF(ar, (size_t)BN * C * HW1);
  float* a2    = allocF(ar, (size_t)BN * C * HW1);
  _Float16* panel = allocH(ar, (size_t)MSTRIP * 128 * 49);  // max K = 6272

  int tot;

  // ---- pyramid ----
  tot = 2 * T * C * HW2;
  pool2x2_k<<<cdiv(tot, 256), 256, 0, stream>>>(X, X2, H1, W1, tot);
  tot = 2 * T * C * HW3;
  pool2x2_k<<<cdiv(tot, 256), 256, 0, stream>>>(X2, X3, H2, W2, tot);

  tot = BN * C * HW1;
  gather_k<<<cdiv(tot, 256), 256, 0, stream>>>(X,  n1, c1, T, tc, C * HW1, tot);
  tot = BN * C * HW2;
  gather_k<<<cdiv(tot, 256), 256, 0, stream>>>(X2, n2, c2, T, tc, C * HW2, tot);
  tot = BN * C * HW3;
  gather_k<<<cdiv(tot, 256), 256, 0, stream>>>(X3, n3, c3, T, tc, C * HW3, tot);

  // ---- level 3 ----
  offset_branch(stream, n3, c3, H3, W3, 3, 1, P_o3cw, o3_cb, P_o3d1ow, o3_d1ob,
                P_o3d1w, P_o3d2ow, o3_d2ob, P_o3d2w, off3, tmpH, tmpH2, tmpOff, panel);
  tot = BN * 144 * HW2;
  upsample2_k<<<cdiv(tot, 256), 256, 0, stream>>>(off3, off3u, H3, W3, 2.0f, tot);

  // ---- level 2 ----
  dconv_layer(stream, n2, off3u, H2, W2, P_al2, 64, z2, panel);
  offset_branch(stream, z2, c2, H2, W2, 5, 2, P_o2cw, o2_cb, P_o2d1ow, o2_d1ob,
                P_o2d1w, P_o2d2ow, o2_d2ob, P_o2d2w, off2, tmpH, tmpH2, tmpOff, panel);
  tot = BN * 144 * HW2;
  add_k<<<cdiv(tot, 256), 256, 0, stream>>>(off2, off3u, tot);
  tot = BN * 144 * HW1;
  upsample2_k<<<cdiv(tot, 256), 256, 0, stream>>>(off2, off2u, H2, W2, 2.0f, tot);

  // ---- level 1 ----
  dconv_layer(stream, n1, off2u, H1, W1, P_al1, 64, z1, panel);
  offset_branch(stream, z1, c1, H1, W1, 7, 3, P_o1cw, o1_cb, P_o1d1ow, o1_d1ob,
                P_o1d1w, P_o1d2ow, o1_d2ob, P_o1d2w, off1, tmpH, tmpH2, tmpOff, panel);
  tot = BN * 144 * HW1;
  add_k<<<cdiv(tot, 256), 256, 0, stream>>>(off1, off2u, tot);

  // ---- final alignment + post dcp ----
  dconv_layer(stream, n1, off1, H1, W1, P_fi, 64, a1, panel);
  conv_layer(stream, a1, nullptr, 64, 64, H1, W1, 3, 1, P_postow, post_ob, 144,
             tmpOff, panel);
  dconv_layer(stream, a1, tmpOff, H1, W1, P_postw, 64, a2, panel);

  // ---- temporal re-assembly ----
  tot = 2 * T * C * HW1;
  assemble_k<<<cdiv(tot, 256), 256, 0, stream>>>(X, a2, (float*)d_out, T, tc,
                                                 C * HW1, tot);
}
